// LSTM_35751307772703
// MI455X (gfx1250) — compile-verified, matrix operand order checked by reading, not run on hardware
//
#include <hip/hip_runtime.h>
#include <hip/hip_bf16.h>

// LSTM: B=64, T=512, D=512, H=1024.  z = [x_t; h] @ [Wi; Wh] + b  (K=1536, N=4096)
// Persistent kernel: 128 WGs, each owns 8 h-columns -> 32 z-columns (4 gates x 8).
// Weight slab (1536x32) lives in LDS in pre-swizzled WMMA B-fragment layout (96 KB).
// h double-buffered in global (bf16), device-wide per-step barrier via arrive[t].

#define B_ 64
#define T_ 512
#define D_ 512
#define H_ 1024
#define NWG 128
#define NTHREADS 256
#define SLAB_ELEMS (48 * 2 * 32 * 16) // 49152 bf16 = 96 KB

typedef __bf16 bf16;
typedef bf16 v16bf __attribute__((ext_vector_type(16)));
typedef float v8f __attribute__((ext_vector_type(8)));

union Frag16 {
    v16bf v;
    unsigned short s[16];
    uint4 q[2];
};

__device__ __forceinline__ unsigned short f2bf(float f) {
    unsigned int u = __float_as_uint(f);
    u += 0x7fffu + ((u >> 16) & 1u); // round-to-nearest-even
    return (unsigned short)(u >> 16);
}

__device__ __forceinline__ float sigmoidf_(float x) {
    return 1.0f / (1.0f + __expf(-x));
}

// Zero h double-buffer and per-step arrive counters (re-run every call -> replay-safe).
__global__ void lstm_init_kernel(unsigned short* __restrict__ hbuf, int* __restrict__ arrive) {
    int i = blockIdx.x * blockDim.x + threadIdx.x;
    if (i < 2 * B_ * H_) hbuf[i] = 0; // bf16 +0.0
    if (i < T_) arrive[i] = 0;
}

// x (B,T,D) fp32  ->  xbf (T,B,D) bf16  (time-major so each step's A rows are contiguous)
__global__ void lstm_cvt_x_kernel(const float* __restrict__ x, unsigned short* __restrict__ xbf) {
    size_t i = (size_t)blockIdx.x * blockDim.x + threadIdx.x; // 16,777,216 elems
    int k = (int)(i & 511);
    int t = (int)((i >> 9) & 511);
    int b = (int)(i >> 18);
    xbf[(((size_t)t * B_ + b) << 9) + k] = f2bf(x[i]);
}

__global__ void __launch_bounds__(NTHREADS, 1)
lstm_persistent_kernel(const unsigned short* __restrict__ xbf,
                       const float* __restrict__ Wi,
                       const float* __restrict__ Wh,
                       const float* __restrict__ bias,
                       unsigned short* __restrict__ hbuf,
                       int* __restrict__ arrive,
                       float* __restrict__ out) {
    extern __shared__ unsigned char smem[];
    unsigned short* slabB = (unsigned short*)smem;                       // 96 KB, B-fragment layout
    float* z_ls = (float*)(smem + SLAB_ELEMS * sizeof(unsigned short));  // 64x32 fp32 = 8 KB
    float* c_ls = z_ls + B_ * 32;                                        // 64x8  fp32 = 2 KB

    const int tid  = threadIdx.x;
    const int lane = tid & 31;
    const int wave = tid >> 5;   // 8 waves
    const int mt   = wave & 3;   // M tile 0..3 (rows mt*16..)
    const int nt   = wave >> 2;  // N tile 0..1 (z-cols nt*16..)
    const int wg8  = blockIdx.x * 8; // first owned h-column

    // ---- Stage weight slab into LDS, pre-swizzled into the 16-bit B-matrix (32x16) layout ----
    // B frag layout (mirror of A): lane l holds col n=l&15; halves 0..7 = K kb..kb+7,
    // halves 8..15 = K kb+16..kb+23, kb = (l>>4)*8.
    for (int idx = tid; idx < SLAB_ELEMS; idx += NTHREADS) {
        int h   = idx & 15;
        int ln  = (idx >> 4) & 31;
        int fr  = idx >> 9;   // fragment id 0..95
        int kt  = fr >> 1;    // K tile 0..47
        int ntl = fr & 1;     // N tile 0..1
        int kb  = (ln >> 4) * 8;
        int hh  = (h < 8) ? (kb + h) : (kb + 8 + h); // kb+16+(h-8)
        int k   = kt * 32 + hh;                       // 0..1535
        int zc  = ntl * 16 + (ln & 15);               // z-col within WG, 0..31
        int wcol = (zc >> 3) * H_ + wg8 + (zc & 7);   // gate*H + j
        float wv = (k < D_) ? Wi[(size_t)k * (4 * H_) + wcol]
                            : Wh[(size_t)(k - D_) * (4 * H_) + wcol];
        slabB[idx] = f2bf(wv);
    }
    for (int i = tid; i < B_ * 8; i += NTHREADS) c_ls[i] = 0.0f;

    // Per-lane bias for this wave's output columns (same for all 8 rows of the acc).
    const int zc_l = nt * 16 + (lane & 15);
    const float biasv = bias[(zc_l >> 3) * H_ + wg8 + (zc_l & 7)];

    // A-fragment addressing (16-bit A 16x32 layout): lane holds row = lane&15 (+mt*16),
    // halves 0..7 = K kb..kb+7, halves 8..15 = K kb+16..kb+23, kb=(lane>>4)*8.
    const int arow = mt * 16 + (lane & 15);
    const int kb   = (lane >> 4) * 8;
    const unsigned short* bbase = slabB + (size_t)(nt * 32 + lane) * 16;

    __syncthreads();

    int cur = 0;
    for (int t = 0; t < T_; ++t) {
        const unsigned short* Xt = xbf + (size_t)t * (B_ * D_);
        const unsigned short* Hc = hbuf + (size_t)cur * (B_ * H_);
        unsigned short* Hn       = hbuf + (size_t)(cur ^ 1) * (B_ * H_);

        v8f acc;
#pragma unroll
        for (int i = 0; i < 8; ++i) acc[i] = biasv;

        // ---- K over x_t (K tiles 0..15) ----
        const unsigned short* ap = Xt + arow * D_ + kb;
#pragma unroll 4
        for (int kt = 0; kt < 16; ++kt) {
            Frag16 fa, fb;
            fa.q[0] = *(const uint4*)(ap + kt * 32);
            fa.q[1] = *(const uint4*)(ap + kt * 32 + 16);
            const uint4* bq = (const uint4*)(bbase + (size_t)kt * 1024);
            fb.q[0] = bq[0];
            fb.q[1] = bq[1];
            acc = __builtin_amdgcn_wmma_f32_16x16x32_bf16(false, fa.v, false, fb.v,
                                                          (short)0, acc, false, false);
        }
        // ---- K over h (K tiles 16..47) ----
        const unsigned short* hp = Hc + arow * H_ + kb;
#pragma unroll 4
        for (int kt = 0; kt < 32; ++kt) {
            Frag16 fa, fb;
            fa.q[0] = *(const uint4*)(hp + kt * 32);
            fa.q[1] = *(const uint4*)(hp + kt * 32 + 16);
            const uint4* bq = (const uint4*)(bbase + (size_t)(kt + 16) * 1024);
            fb.q[0] = bq[0];
            fb.q[1] = bq[1];
            acc = __builtin_amdgcn_wmma_f32_16x16x32_bf16(false, fa.v, false, fb.v,
                                                          (short)0, acc, false, false);
        }

        // Prefetch next step's x rows into cache (gfx1250 global_prefetch_b8).
        if (t + 1 < T_) __builtin_prefetch((const void*)(ap + B_ * D_), 0, 0);

        // ---- Scatter z tile to LDS (C/D layout: col = lane&15, row = (lane>>4)*8 + i) ----
        {
            int col = nt * 16 + (lane & 15);
            int rb  = mt * 16 + (lane >> 4) * 8;
#pragma unroll
            for (int i = 0; i < 8; ++i) z_ls[(rb + i) * 32 + col] = acc[i];
        }
        __syncthreads();

        // ---- Gates + state update: 512 (b,j) elements, 2 per thread ----
#pragma unroll
        for (int r = 0; r < 2; ++r) {
            int idx = tid + r * NTHREADS;
            int bb  = idx >> 3;
            int jj  = idx & 7;
            float iz = sigmoidf_(z_ls[bb * 32 + jj]);
            float fz = sigmoidf_(z_ls[bb * 32 + 8 + jj]);
            float gz = tanhf(z_ls[bb * 32 + 16 + jj]);
            float oz = sigmoidf_(z_ls[bb * 32 + 24 + jj]);
            float cn = fz * c_ls[idx] + iz * gz;
            c_ls[idx] = cn;
            float hn = oz * tanhf(cn);
            Hn[bb * H_ + wg8 + jj] = f2bf(hn);
            out[(size_t)bb * (T_ * H_) + (size_t)t * H_ + wg8 + jj] = hn;
        }
        __threadfence();
        __syncthreads();

        // ---- Device-wide step barrier (per-step counter; reset each call by init kernel) ----
        if (tid == 0) {
            __hip_atomic_fetch_add(&arrive[t], 1, __ATOMIC_RELEASE, __HIP_MEMORY_SCOPE_AGENT);
            while (__hip_atomic_load(&arrive[t], __ATOMIC_ACQUIRE, __HIP_MEMORY_SCOPE_AGENT) < NWG)
                __builtin_amdgcn_s_sleep(2);
        }
        __syncthreads();
        cur ^= 1;
    }
}

extern "C" void kernel_launch(void* const* d_in, const int* in_sizes, int n_in,
                              void* d_out, int out_size, void* d_ws, size_t ws_size,
                              hipStream_t stream) {
    const float* x    = (const float*)d_in[0]; // (B,T,D)
    const float* Wi   = (const float*)d_in[1]; // (D,4H)
    const float* Wh   = (const float*)d_in[2]; // (H,4H)
    const float* bias = (const float*)d_in[3]; // (4H,)
    float* out = (float*)d_out;                // (B,T,H) fp32

    // Workspace layout
    unsigned short* xbf  = (unsigned short*)d_ws;                 // T*B*D bf16 = 32 MB
    unsigned short* hbuf = xbf + (size_t)T_ * B_ * D_;            // 2*B*H bf16 = 256 KB
    int* arrive          = (int*)(hbuf + (size_t)2 * B_ * H_);    // T ints

    lstm_init_kernel<<<(2 * B_ * H_ + NTHREADS - 1) / NTHREADS, NTHREADS, 0, stream>>>(hbuf, arrive);

    size_t nx = (size_t)B_ * T_ * D_;
    lstm_cvt_x_kernel<<<(unsigned)(nx / NTHREADS), NTHREADS, 0, stream>>>(x, xbf);

    size_t lds_bytes = SLAB_ELEMS * sizeof(unsigned short) // 96 KB weight slab
                     + (size_t)B_ * 32 * sizeof(float)     // 8 KB z tile
                     + (size_t)B_ * 8 * sizeof(float);     // 2 KB cell state
    lstm_persistent_kernel<<<NWG, NTHREADS, lds_bytes, stream>>>(xbf, Wi, Wh, bias, hbuf,
                                                                 arrive, out);
}
